// BNN_65360812310623
// MI455X (gfx1250) — 8-bit matrix operand order repaired and checked by reading, compiled, not run on hardware
//
#include <hip/hip_runtime.h>
#include <hip/hip_bf16.h>
#include <stdint.h>


typedef __attribute__((ext_vector_type(16))) _Float16 v16h;
typedef __attribute__((ext_vector_type(8)))  float    v8f;
typedef __attribute__((ext_vector_type(16))) int      v16i;
typedef int v4i_vs __attribute__((__vector_size__(16)));
typedef __attribute__((address_space(1))) v4i_vs g_v4i;
typedef __attribute__((address_space(3))) v4i_vs l_v4i;

#define DEV __device__ __forceinline__

static constexpr int BATCH = 16384;
static constexpr int K1    = 784;
static constexpr int K1P   = 800;
static constexpr int H     = 1024;
static constexpr int LDSB  = 144;

#if __has_builtin(__builtin_amdgcn_global_load_async_to_lds_b128)
#define USE_ASYNC_LDS 1
#else
#define USE_ASYNC_LDS 0
#endif

DEV void wait_async_le2() {
#if __has_builtin(__builtin_amdgcn_s_wait_asynccnt)
  __builtin_amdgcn_s_wait_asynccnt(2);
#else
  asm volatile("s_wait_asynccnt 0x2" ::: "memory");
#endif
}
DEV void wait_async_le0() {
#if __has_builtin(__builtin_amdgcn_s_wait_asynccnt)
  __builtin_amdgcn_s_wait_asynccnt(0);
#else
  asm volatile("s_wait_asynccnt 0x0" ::: "memory");
#endif
}

union FragH  { v16h v; int4 q[2]; };
union FragA8 { v16i v; unsigned long long u[8]; };
union FragB8 { v16i v; int4 q[4]; };

DEV FragH load_h16(const _Float16* base, int row0, int ld, int kc, int lane) {
  FragH f;
  const _Float16* p = base + (size_t)(row0 + (lane & 15)) * ld + kc + ((lane >> 4) << 3);
  f.q[0] = *(const int4*)(p);
  f.q[1] = *(const int4*)(p + 16);
  return f;
}

DEV FragA8 load_a8(const uint8_t* base, int row0, int ld, int kc, int lane) {
  FragA8 f;
  const uint8_t* p = base + (size_t)(row0 + (lane & 15)) * ld + kc + ((lane >> 4) << 4);
#pragma unroll
  for (int h = 0; h < 2; ++h) {
    const uint8_t* ph = p + h * 64;
    f.u[h * 4 + 0] = *(const unsigned long long*)(ph + 0);
    f.u[h * 4 + 1] = *(const unsigned long long*)(ph + 8);
    f.u[h * 4 + 2] = *(const unsigned long long*)(ph + 32);
    f.u[h * 4 + 3] = *(const unsigned long long*)(ph + 40);
  }
  return f;
}

DEV FragB8 load_b8(const uint8_t* base, int row0, int ld, int kc, int lane) {
  FragB8 f;
  const uint8_t* p = base + (size_t)(row0 + (lane & 15)) * ld + kc + ((lane >> 4) << 4);
  f.q[0] = *(const int4*)(p + 0);
  f.q[1] = *(const int4*)(p + 32);
  f.q[2] = *(const int4*)(p + 64);
  f.q[3] = *(const int4*)(p + 96);
  return f;
}

DEV FragB8 load_b8_lds(const uint8_t* sB, int n0, int lane) {
  FragB8 f;
  const uint8_t* p = sB + (n0 + (lane & 15)) * LDSB + ((lane >> 4) << 4);
  f.q[0] = *(const int4*)(p + 0);
  f.q[1] = *(const int4*)(p + 32);
  f.q[2] = *(const int4*)(p + 64);
  f.q[3] = *(const int4*)(p + 96);
  return f;
}

DEV v8f vzero() { v8f z = {0.f, 0.f, 0.f, 0.f, 0.f, 0.f, 0.f, 0.f}; return z; }
DEV uint8_t fp8_sign(float w) { return w > 0.f ? 0x38 : (w < 0.f ? 0xB8 : 0x00); }


__global__ void k_prep_x(const float* __restrict__ x,
                         _Float16* __restrict__ xhi, _Float16* __restrict__ xlo) {
  const size_t N = (size_t)BATCH * K1P;
  for (size_t i = (size_t)blockIdx.x * blockDim.x + threadIdx.x; i < N;
       i += (size_t)gridDim.x * blockDim.x) {
    int r = (int)(i / K1P), k = (int)(i % K1P);
    float f = (k < K1) ? x[(size_t)r * K1 + k] : 0.f;
    _Float16 hi = (_Float16)f;
    _Float16 lo = (_Float16)(f - (float)hi);
    xhi[i] = hi; xlo[i] = lo;
  }
}

__global__ void k_prep_w1(const float* __restrict__ W1, _Float16* __restrict__ w1h) {
  const size_t N = (size_t)H * K1P;
  for (size_t i = (size_t)blockIdx.x * blockDim.x + threadIdx.x; i < N;
       i += (size_t)gridDim.x * blockDim.x) {
    int r = (int)(i / K1P), k = (int)(i % K1P);
    float s = 0.f;
    if (k < K1) { float w = W1[(size_t)r * K1 + k]; s = w > 0.f ? 1.f : (w < 0.f ? -1.f : 0.f); }
    w1h[i] = (_Float16)s;
  }
}

__global__ void k_prep_sgn8(const float* __restrict__ w, uint8_t* __restrict__ o, int n) {
  for (size_t i = (size_t)blockIdx.x * blockDim.x + threadIdx.x; i < (size_t)n;
       i += (size_t)gridDim.x * blockDim.x)
    o[i] = fp8_sign(w[i]);
}

__global__ void k_prep_w4(const float* __restrict__ W4, uint8_t* __restrict__ o) {
  for (size_t i = (size_t)blockIdx.x * blockDim.x + threadIdx.x; i < (size_t)16 * H;
       i += (size_t)gridDim.x * blockDim.x) {
    int r = (int)(i >> 10), k = (int)(i & (H - 1));
    o[i] = (r < 10) ? fp8_sign(W4[(size_t)r * H + k]) : (uint8_t)0;
  }
}

__global__ void k_zero_i32(int* __restrict__ p, int n) {
  int i = blockIdx.x * blockDim.x + threadIdx.x;
  if (i < n) p[i] = 0;
}


__global__ __launch_bounds__(256) void k_gemm1(const _Float16* __restrict__ xhi,
                                               const _Float16* __restrict__ xlo,
                                               const _Float16* __restrict__ w1h,
                                               float* __restrict__ h1) {
  const int lane = threadIdx.x & 31, wave = threadIdx.x >> 5;
  const int mW = blockIdx.x * 128 + wave * 16;
  const int nB = blockIdx.y * 64;
  v8f acc[4];
#pragma unroll
  for (int t = 0; t < 4; ++t) acc[t] = vzero();

  for (int kc = 0; kc < K1P; kc += 32) {
    FragH ahi = load_h16(xhi, mW, K1P, kc, lane);
    FragH alo = load_h16(xlo, mW, K1P, kc, lane);
#pragma unroll
    for (int t = 0; t < 4; ++t) {
      FragH b = load_h16(w1h, nB + t * 16, K1P, kc, lane);
      acc[t] = __builtin_amdgcn_wmma_f32_16x16x32_f16(false, ahi.v, false, b.v,
                                                      (short)0, acc[t], false, false);
      acc[t] = __builtin_amdgcn_wmma_f32_16x16x32_f16(false, alo.v, false, b.v,
                                                      (short)0, acc[t], false, false);
    }
  }
#pragma unroll
  for (int t = 0; t < 4; ++t)
#pragma unroll
    for (int r = 0; r < 8; ++r) {
      int m = mW + r + ((lane >> 4) << 3);
      int n = nB + t * 16 + (lane & 15);
      h1[(size_t)m * H + n] = acc[t][r];
    }
}


__global__ void k_colstatA(const float* __restrict__ h1, float* __restrict__ part) {
  const int rg = blockIdx.x, col = blockIdx.y * 256 + threadIdx.x;
  const int r0 = rg * 256;
  float s = 0.f, q = 0.f;
  for (int i = 0; i < 256; ++i) {
    float v = h1[(size_t)(r0 + i) * H + col];
    s += v; q += v * v;
  }
  float2 o; o.x = s; o.y = q;
  *(float2*)(part + ((size_t)rg * H + col) * 2) = o;
}

__global__ void k_colstatB(const float* __restrict__ part,
                           float* __restrict__ mu, float* __restrict__ rs) {
  const int col = blockIdx.x * 256 + threadIdx.x;
  float s = 0.f, q = 0.f;
  for (int rg = 0; rg < 64; ++rg) {
    float2 p = *(const float2*)(part + ((size_t)rg * H + col) * 2);
    s += p.x; q += p.y;
  }
  float m = s * (1.0f / BATCH);
  float var = q * (1.0f / BATCH) - m * m;
  mu[col] = m;
  rs[col] = rsqrtf(var + 1e-5f);
}

__global__ void k_bin1(const float* __restrict__ h1, const float* __restrict__ mu,
                       const float* __restrict__ rs, const float* __restrict__ g,
                       const float* __restrict__ b, uint8_t* __restrict__ a) {
  const size_t N = (size_t)BATCH * H;
  for (size_t i = (size_t)blockIdx.x * blockDim.x + threadIdx.x; i < N;
       i += (size_t)gridDim.x * blockDim.x) {
    int c = (int)(i & (H - 1));
    float val = (h1[i] - mu[c]) * rs[c] * g[c] + b[c];
    a[i] = fp8_sign(val);
  }
}


#if USE_ASYNC_LDS
DEV void stage_b(const uint8_t* __restrict__ W, uint8_t* sB, int nB, int kc, int tid) {
  int seg = tid;
#pragma unroll
  for (int p = 0; p < 2; ++p, seg += 256) {
    int row = seg >> 3;
    int o16 = (seg & 7) << 4;
    const uint8_t* gp = W + (size_t)(nB + row) * H + kc + o16;
    uint8_t* lp = sB + row * LDSB + o16;
    __builtin_amdgcn_global_load_async_to_lds_b128((g_v4i*)gp, (l_v4i*)lp, 0, 0);
  }
}
#endif

__global__ __launch_bounds__(256) void k_gemm_noisy(const uint8_t* __restrict__ A,
                                                    const uint8_t* __restrict__ W,
                                                    const float* __restrict__ U,
                                                    uint8_t* __restrict__ Vout,
                                                    int* __restrict__ colsum) {
  __shared__ int s_cs[64];
#if USE_ASYNC_LDS
  __shared__ __align__(16) uint8_t sB[2][64 * LDSB];
#endif
  const int tid = threadIdx.x, lane = tid & 31, wave = tid >> 5;
  if (tid < 64) s_cs[tid] = 0;

  const int mW = blockIdx.x * 128 + wave * 16;
  const int nB = blockIdx.y * 64;
  v8f acc[4];
#pragma unroll
  for (int t = 0; t < 4; ++t) acc[t] = vzero();

#if USE_ASYNC_LDS
  stage_b(W, &sB[0][0], nB, 0, tid);
  int buf = 0;
  for (int kc = 0; kc < H; kc += 128) {
    const bool more = (kc + 128 < H);
    if (more) stage_b(W, &sB[buf ^ 1][0], nB, kc + 128, tid);
    if (more) wait_async_le2(); else wait_async_le0();
    __syncthreads();
    FragA8 a = load_a8(A, mW, H, kc, lane);
    if (more)
      __builtin_prefetch(A + (size_t)(mW + (lane & 15)) * H + kc + 128, 0, 1);
#pragma unroll
    for (int t = 0; t < 4; ++t) {
      FragB8 b = load_b8_lds(&sB[buf][0], t * 16, lane);
      acc[t] = __builtin_amdgcn_wmma_f32_16x16x128_fp8_fp8(a.v, b.v, (short)0,
                                                           acc[t], false, false);
    }
    __syncthreads();
    buf ^= 1;
  }
#else
  __syncthreads();
  for (int kc = 0; kc < H; kc += 128) {
    FragA8 a = load_a8(A, mW, H, kc, lane);
    if (kc + 128 < H)
      __builtin_prefetch(A + (size_t)(mW + (lane & 15)) * H + kc + 128, 0, 1);
#pragma unroll
    for (int t = 0; t < 4; ++t) {
      FragB8 b = load_b8(W, nB + t * 16, H, kc, lane);
      acc[t] = __builtin_amdgcn_wmma_f32_16x16x128_fp8_fp8(a.v, b.v, (short)0,
                                                           acc[t], false, false);
    }
  }
#endif

#pragma unroll
  for (int t = 0; t < 4; ++t)
#pragma unroll
    for (int r = 0; r < 8; ++r) {
      int m = mW + r + ((lane >> 4) << 3);
      int n = nB + t * 16 + (lane & 15);
      float h = acc[t][r];
      float u = U[(size_t)m * H + n];
      float s = h > 0.f ? 1.f : -1.f;
      float prob = 0.5f * __expf(-h * h * 0.02f);
      bool flip = (u < prob) && (fabsf(h) <= 50.f);
      float v = flip ? -s : s;
      Vout[(size_t)m * H + n] = v > 0.f ? 0x38 : 0xB8;
      atomicAdd(&s_cs[n - nB], v > 0.f ? 1 : -1);
    }
  __syncthreads();
  if (tid < 64) atomicAdd(&colsum[nB + tid], s_cs[tid]);
}

__global__ void k_stats_pm1(const int* __restrict__ cs, const float* __restrict__ g,
                            const float* __restrict__ b,
                            uint8_t* __restrict__ lutP, uint8_t* __restrict__ lutM) {
  int n = blockIdx.x * blockDim.x + threadIdx.x;
  if (n < H) {
    float mu = (float)cs[n] * (1.0f / BATCH);
    float rs = rsqrtf(1.0f - mu * mu + 1e-5f);
    lutP[n] = fp8_sign((1.0f - mu) * rs * g[n] + b[n]);
    lutM[n] = fp8_sign((-1.0f - mu) * rs * g[n] + b[n]);
  }
}

__global__ void k_bin_lut(const uint8_t* __restrict__ v, const uint8_t* __restrict__ lutP,
                          const uint8_t* __restrict__ lutM, uint8_t* __restrict__ a) {
  const size_t N = (size_t)BATCH * H;
  for (size_t i = (size_t)blockIdx.x * blockDim.x + threadIdx.x; i < N;
       i += (size_t)gridDim.x * blockDim.x)
    a[i] = (v[i] == 0x38) ? lutP[i & (H - 1)] : lutM[i & (H - 1)];
}


__global__ __launch_bounds__(256) void k_gemm4(const uint8_t* __restrict__ A,
                                               const uint8_t* __restrict__ W4b,
                                               float* __restrict__ out) {
  const int lane = threadIdx.x & 31, wave = threadIdx.x >> 5;
  const int mW = blockIdx.x * 128 + wave * 16;
  v8f acc = vzero();
  for (int kc = 0; kc < H; kc += 128) {
    FragA8 a = load_a8(A, mW, H, kc, lane);
    FragB8 b = load_b8(W4b, 0, H, kc, lane);
    acc = __builtin_amdgcn_wmma_f32_16x16x128_fp8_fp8(a.v, b.v, (short)0, acc, false, false);
  }
  const int n = lane & 15;
#pragma unroll
  for (int r = 0; r < 8; ++r) {
    int m = mW + r + ((lane >> 4) << 3);
    if (n < 10) out[(size_t)m * 10 + n] = acc[r];
  }
}


extern "C" void kernel_launch(void* const* d_in, const int* in_sizes, int n_in,
                              void* d_out, int out_size, void* d_ws, size_t ws_size,
                              hipStream_t stream) {
  (void)in_sizes; (void)n_in; (void)out_size; (void)ws_size;
  const float* x  = (const float*)d_in[0];
  const float* u2 = (const float*)d_in[1];
  const float* u3 = (const float*)d_in[2];
  const float* W1 = (const float*)d_in[3];
  const float* W2 = (const float*)d_in[4];
  const float* W3 = (const float*)d_in[5];
  const float* W4 = (const float*)d_in[6];
  const float* g1 = (const float*)d_in[7];  const float* b1 = (const float*)d_in[8];
  const float* g2 = (const float*)d_in[9];  const float* b2 = (const float*)d_in[10];
  const float* g3 = (const float*)d_in[11]; const float* b3 = (const float*)d_in[12];

  char* ws = (char*)d_ws;
  constexpr size_t XHI = 0;
  constexpr size_t XLO = 26214400;
  constexpr size_t W1H = 52428800;
  constexpr size_t W2B = 54067200;
  constexpr size_t W3B = 55115776;
  constexpr size_t W4B = 56164352;
  constexpr size_t H1  = 56180736;
  constexpr size_t A1  = XHI;
  constexpr size_t V2  = H1;
  constexpr size_t A2  = H1 + 16777216;
  constexpr size_t V3  = H1 + 33554432;
  constexpr size_t A3  = H1 + 50331648;
  constexpr size_t PART = 123289600;
  constexpr size_t MU1 = 123813888, RS1 = 123817984;
  constexpr size_t CS2 = 123822080, CS3 = 123826176;
  constexpr size_t LP2 = 123830272, LM2 = 123831296;
  constexpr size_t LP3 = 123832320, LM3 = 123833344;

  _Float16* xhi = (_Float16*)(ws + XHI);
  _Float16* xlo = (_Float16*)(ws + XLO);
  _Float16* w1h = (_Float16*)(ws + W1H);
  uint8_t *w2b = (uint8_t*)(ws + W2B), *w3b = (uint8_t*)(ws + W3B), *w4b = (uint8_t*)(ws + W4B);
  float* h1 = (float*)(ws + H1);
  uint8_t *a1 = (uint8_t*)(ws + A1), *v2 = (uint8_t*)(ws + V2), *a2 = (uint8_t*)(ws + A2);
  uint8_t *v3 = (uint8_t*)(ws + V3), *a3 = (uint8_t*)(ws + A3);
  float* part = (float*)(ws + PART);
  float *mu1 = (float*)(ws + MU1), *rs1 = (float*)(ws + RS1);
  int *cs2 = (int*)(ws + CS2), *cs3 = (int*)(ws + CS3);
  uint8_t *lp2 = (uint8_t*)(ws + LP2), *lm2 = (uint8_t*)(ws + LM2);
  uint8_t *lp3 = (uint8_t*)(ws + LP3), *lm3 = (uint8_t*)(ws + LM3);

  const dim3 blk(256);
  k_prep_x   <<<dim3(8192), blk, 0, stream>>>(x, xhi, xlo);
  k_prep_w1  <<<dim3(3200), blk, 0, stream>>>(W1, w1h);
  k_prep_sgn8<<<dim3(4096), blk, 0, stream>>>(W2, w2b, H * H);
  k_prep_sgn8<<<dim3(4096), blk, 0, stream>>>(W3, w3b, H * H);
  k_prep_w4  <<<dim3(64),   blk, 0, stream>>>(W4, w4b);
  k_zero_i32 <<<dim3(8),    blk, 0, stream>>>(cs2, 2048);
  k_gemm1   <<<dim3(128, 16), blk, 0, stream>>>(xhi, xlo, w1h, h1);
  k_colstatA<<<dim3(64, 4),   blk, 0, stream>>>(h1, part);
  k_colstatB<<<dim3(4),       blk, 0, stream>>>(part, mu1, rs1);
  k_bin1    <<<dim3(8192),    blk, 0, stream>>>(h1, mu1, rs1, g1, b1, a1);
  k_gemm_noisy<<<dim3(128, 16), blk, 0, stream>>>(a1, w2b, u2, v2, cs2);
  k_stats_pm1 <<<dim3(4),       blk, 0, stream>>>(cs2, g2, b2, lp2, lm2);
  k_bin_lut   <<<dim3(8192),    blk, 0, stream>>>(v2, lp2, lm2, a2);
  k_gemm_noisy<<<dim3(128, 16), blk, 0, stream>>>(a2, w3b, u3, v3, cs3);
  k_stats_pm1 <<<dim3(4),       blk, 0, stream>>>(cs3, g3, b3, lp3, lm3);
  k_bin_lut   <<<dim3(8192),    blk, 0, stream>>>(v3, lp3, lm3, a3);
  k_gemm4<<<dim3(128), blk, 0, stream>>>(a3, w4b, (float*)d_out);
}
